// Gemma4VisionAttention_33655363731985
// MI455X (gfx1250) — matrix operand order repaired and checked by reading, not compiled, not run on hardware
//
#include <hip/hip_runtime.h>

// Gemma4 vision attention for gfx1250 (MI455X), wave32 + WMMA bf16 + async-LDS.
// B=4, S=1024, HID=1024, H=16, HK=8, D=64, THETA=10000, EPS=1e-6, scale=1.
//
// Pipeline:
//   1) gemm_wmma<float>  x3 : hidden @ {Wq,Wk,Wv} -> qkv fp32 (4096 x 2048)
//   2) norm_rope            : per-head RMS norm (+gamma q/k), 2D RoPE q/k,
//                             writes Q/K/V bf16 head-major
//   3) attn_kernel          : flash attention; K tile staged with
//                             global_load_async_to_lds_b128 (ASYNCcnt),
//                             QK^T and P@V via v_wmma_f32_16x16x32_bf16,
//                             softmax fully in registers (shfl within halves)
//   4) gemm_wmma<__bf16>    : attn_out @ Wo -> d_out fp32
//
// Workspace (d_ws): [0,32MB) qkv f32 | [32,40) Qb | [40,44) Kb | [44,48) Vb
//                   | [48,56) attn_out   (bf16 buffers)

typedef __attribute__((ext_vector_type(16))) __bf16 v16bf;
typedef __attribute__((ext_vector_type(8)))  float  v8f;

__device__ __forceinline__ __bf16 f2bf(float f) {
  union { float f; unsigned u; } a; a.f = f;
  unsigned r = a.u + 0x7FFFu + ((a.u >> 16) & 1u);   // round-to-nearest-even
  union { unsigned short s; __bf16 b; } c; c.s = (unsigned short)(r >> 16);
  return c.b;
}
__device__ __forceinline__ unsigned pack_bf16(float lo, float hi) {
  union { float f; unsigned u; } a, b; a.f = lo; b.f = hi;
  unsigned rl = (a.u + 0x7FFFu + ((a.u >> 16) & 1u)) >> 16;
  unsigned rh = (b.u + 0x7FFFu + ((b.u >> 16) & 1u)) & 0xFFFF0000u;
  return rh | rl;
}

__device__ __forceinline__ v8f wmma_bf16(v16bf a, v16bf b, v8f c) {
  return __builtin_amdgcn_wmma_f32_16x16x32_bf16(
      false, a, false, b, (short)0, c, false, false);
}

// Async 16-byte global -> LDS copy (ASYNCcnt-tracked DMA path).
// Generic shared pointers carry the LDS byte offset in their low 32 bits.
__device__ __forceinline__ void async_b128(void* ldsDst, const void* gSrc) {
  asm volatile("global_load_async_to_lds_b128 %0, %1, off"
               :: "v"((unsigned)(size_t)ldsDst),
                  "v"((unsigned long long)(size_t)gSrc)
               : "memory");
}

// A fragment (16x32 bf16): lane L -> row M=L&15, K = kb..kb+15 contiguous,
// kb = (L/16)*16.  (this file's k order, not the ISA table order; load_frag_b matches)
__device__ __forceinline__ v16bf load_frag_a(const __bf16* p, int stride, int k0, int lane) {
  int r  = lane & 15;
  int kb = k0 + ((lane >> 4) << 4);
  const __bf16* q = p + r * stride + kb;
  v16bf a;
#pragma unroll
  for (int e = 0; e < 8; ++e) a[e] = q[e];
#pragma unroll
  for (int e = 0; e < 8; ++e) a[8 + e] = q[8 + e];
  return a;
}

// B fragment (32x16 bf16), N-major LDS (p[n][k]): lane L -> col N=L&15,
// K = (L/16)*16 + 0..15 contiguous.  (same k order as load_frag_a, not the ISA table order)
__device__ __forceinline__ v16bf load_frag_b(const __bf16* p, int stride, int k0, int lane) {
  int n  = lane & 15;
  int kb = k0 + ((lane >> 4) << 4);
  const __bf16* q = p + n * stride + kb;
  v16bf a;
#pragma unroll
  for (int e = 0; e < 16; ++e) a[e] = q[e];
  return a;
}

// Stage 16 contiguous elements into LDS as bf16 via two b128 stores.
__device__ __forceinline__ void stage16(const float* ap, __bf16* dp) {
  const float4* a4 = (const float4*)ap;
  float4 x0 = a4[0], x1 = a4[1], x2 = a4[2], x3 = a4[3];
  uint4 o0, o1;
  o0.x = pack_bf16(x0.x, x0.y); o0.y = pack_bf16(x0.z, x0.w);
  o0.z = pack_bf16(x1.x, x1.y); o0.w = pack_bf16(x1.z, x1.w);
  o1.x = pack_bf16(x2.x, x2.y); o1.y = pack_bf16(x2.z, x2.w);
  o1.z = pack_bf16(x3.x, x3.y); o1.w = pack_bf16(x3.z, x3.w);
  ((uint4*)dp)[0] = o0; ((uint4*)dp)[1] = o1;
}
__device__ __forceinline__ void stage16(const __bf16* ap, __bf16* dp) {
  const uint4* a4 = (const uint4*)ap;
  ((uint4*)dp)[0] = a4[0]; ((uint4*)dp)[1] = a4[1];
}

// ---------------------------------------------------------------------------
// GEMM: C[m, colOff+n] = sum_k A[m,k] * W[k,n];  M,N,K multiples of 64.
// 256 threads = 8 waves; BM=BN=BK=64; wave (wm,wn) 4x2; 2 output tiles/wave.
// ---------------------------------------------------------------------------
template <typename TA>
__global__ __launch_bounds__(256) void gemm_wmma(
    const TA* __restrict__ A, const float* __restrict__ W,
    float* __restrict__ C, int ldc, int colOff, int M, int N, int K) {
  __shared__ __align__(16) __bf16 As[64 * 72];  // [m][k]
  __shared__ __align__(16) __bf16 Bs[64 * 72];  // [n][k]

  const int tid  = threadIdx.x;
  const int lane = tid & 31;
  const int w    = tid >> 5;
  const int wm   = w >> 1, wn = w & 1;
  const int m0   = blockIdx.y * 64;
  const int n0   = blockIdx.x * 64;

  v8f acc0, acc1;
#pragma unroll
  for (int r = 0; r < 8; ++r) { acc0[r] = 0.f; acc1[r] = 0.f; }

  for (int k0 = 0; k0 < K; k0 += 64) {
    __syncthreads();
    {  // stage A tile 64x64 -> bf16, 2 x ds_store_b128 per thread
      int row = tid >> 2, kc = (tid & 3) * 16;
      stage16(A + (size_t)(m0 + row) * K + k0 + kc, As + row * 72 + kc);
      if (k0 + 64 < K)  // -> global_prefetch_b8
        __builtin_prefetch(A + (size_t)(m0 + row) * K + k0 + 64 + kc, 0, 1);
    }
    {  // stage B tile (transpose K-major -> N-major), packed b128 stores
      int n = tid & 63, kb = (tid >> 6) * 16;
      const float* wp = W + (size_t)(k0 + kb) * N + n0 + n;
      uint4 o0, o1;
      o0.x = pack_bf16(wp[0], wp[(size_t)N]);
      o0.y = pack_bf16(wp[(size_t)2 * N], wp[(size_t)3 * N]);
      o0.z = pack_bf16(wp[(size_t)4 * N], wp[(size_t)5 * N]);
      o0.w = pack_bf16(wp[(size_t)6 * N], wp[(size_t)7 * N]);
      o1.x = pack_bf16(wp[(size_t)8 * N], wp[(size_t)9 * N]);
      o1.y = pack_bf16(wp[(size_t)10 * N], wp[(size_t)11 * N]);
      o1.z = pack_bf16(wp[(size_t)12 * N], wp[(size_t)13 * N]);
      o1.w = pack_bf16(wp[(size_t)14 * N], wp[(size_t)15 * N]);
      uint4* dp = (uint4*)(Bs + n * 72 + kb);
      dp[0] = o0; dp[1] = o1;
    }
    __syncthreads();
#pragma unroll
    for (int kk = 0; kk < 2; ++kk) {
      v16bf af = load_frag_a(As + wm * 16 * 72, 72, kk * 32, lane);
      v16bf b0 = load_frag_b(Bs + (wn * 32) * 72, 72, kk * 32, lane);
      acc0 = wmma_bf16(af, b0, acc0);
      v16bf b1 = load_frag_b(Bs + (wn * 32 + 16) * 72, 72, kk * 32, lane);
      acc1 = wmma_bf16(af, b1, acc1);
    }
  }

  const int hi = lane >> 4, nc = lane & 15;
#pragma unroll
  for (int r = 0; r < 8; ++r) {  // C layout: VGPR r -> M=r / r+8
    int row = m0 + wm * 16 + r + hi * 8;
    size_t base = (size_t)row * ldc + colOff + n0 + wn * 32;
    C[base + nc]      = acc0[r];
    C[base + 16 + nc] = acc1[r];
  }
}

// ---------------------------------------------------------------------------
// RMS norm (+gamma) and 2D RoPE.  One wave per (b,s,head)-row of 64 elems.
// lanes 0-15 -> pair (i, i+16), pos.x; lanes 16-31 -> pair (32+i, 48+i), pos.y
// ---------------------------------------------------------------------------
__global__ __launch_bounds__(256) void norm_rope(
    const float* __restrict__ qkv, const float* __restrict__ qg,
    const float* __restrict__ kg, const int* __restrict__ pos,
    __bf16* __restrict__ Qb, __bf16* __restrict__ Kb, __bf16* __restrict__ Vb) {
  const int w = threadIdx.x >> 5, lane = threadIdx.x & 31;
  const long task = (long)blockIdx.x * 8 + w;  // B*S*(H+2*HK) = 131072
  const int bs = (int)(task >> 5);
  const int hh = (int)(task & 31);
  const int b = bs >> 10, s = bs & 1023;

  const int i  = lane & 15;
  const bool hiHalf = lane >= 16;
  const int e0 = hiHalf ? (32 + i) : i;
  const int e1 = e0 + 16;

  const float* row = qkv + (size_t)bs * 2048;
  const float* src;
  const float* gamma = nullptr;
  __bf16* dst;
  bool doRope = true;
  if (hh < 16) {                 // Q head
    src = row + hh * 64;  gamma = qg;
    dst = Qb + ((size_t)(b * 16 + hh) * 1024 + s) * 64;
  } else if (hh < 24) {          // K head
    int kh = hh - 16;
    src = row + 1024 + kh * 64;  gamma = kg;
    dst = Kb + ((size_t)(b * 8 + kh) * 1024 + s) * 64;
  } else {                       // V head (no gamma, no rope)
    int kh = hh - 24;
    src = row + 1536 + kh * 64;
    dst = Vb + ((size_t)(b * 8 + kh) * 1024 + s) * 64;
    doRope = false;
  }

  float x0 = src[e0], x1 = src[e1];
  float ss = x0 * x0 + x1 * x1;
#pragma unroll
  for (int m = 16; m > 0; m >>= 1) ss += __shfl_xor(ss, m, 32);
  float inv = rsqrtf(ss * (1.0f / 64.0f) + 1e-6f);
  float n0v = x0 * inv, n1v = x1 * inv;
  if (gamma) { n0v *= 1.0f + gamma[e0]; n1v *= 1.0f + gamma[e1]; }
  if (doRope) {
    int p = pos[(size_t)bs * 2 + (hiHalf ? 1 : 0)];
    float invf = __powf(10000.0f, -(float)i * (1.0f / 16.0f));
    float ang = (float)p * invf;  // ROPE_SCALE = 1
    float sn, cs;
    __sincosf(ang, &sn, &cs);
    float o0 = n0v * cs - n1v * sn;
    float o1 = n1v * cs + n0v * sn;
    n0v = o0; n1v = o1;
  }
  dst[e0] = f2bf(n0v);
  dst[e1] = f2bf(n1v);
}

// ---------------------------------------------------------------------------
// Flash attention.  grid = (S/64, B*H); block = 128 (4 waves); wave = 16 q-rows.
// K tile async-DMA'd to LDS; V tile transposed to d-major with packed stores.
// Softmax state lives in registers: C-frag VGPR r == row r (lanes 0-15) or
// r+8 (lanes 16-31), so shfl_xor masks 1..8 reduce each row inside its half.
// ---------------------------------------------------------------------------
__global__ __launch_bounds__(128) void attn_kernel(
    const __bf16* __restrict__ Qb, const __bf16* __restrict__ Kb,
    const __bf16* __restrict__ Vb, const unsigned char* __restrict__ mask,
    __bf16* __restrict__ attnOut) {
  __shared__ __align__(16) __bf16   Ks[64 * 72];      // [key][d]
  __shared__ __align__(16) unsigned Vs32[64 * 36];    // [d][key] packed pairs
  __shared__ __align__(16) __bf16   Pb[4][16 * 72];   // per-wave probs 16x64

  const int tid = threadIdx.x, lane = tid & 31, w = tid >> 5;
  const int bh = blockIdx.y, b = bh >> 4, h = bh & 15, kh = h >> 1;  // GQA 2:1
  const int qr0 = blockIdx.x * 64 + w * 16;
  const int hi = lane >> 4, nc = lane & 15;

  const __bf16* Qhead = Qb + (size_t)bh * 1024 * 64;
  const __bf16* Khead = Kb + (size_t)(b * 8 + kh) * 1024 * 64;
  const __bf16* Vhead = Vb + (size_t)(b * 8 + kh) * 1024 * 64;

  v16bf qa0 = load_frag_a(Qhead + (size_t)qr0 * 64, 64, 0, lane);
  v16bf qa1 = load_frag_a(Qhead + (size_t)qr0 * 64, 64, 32, lane);

  v8f O[4];
#pragma unroll
  for (int nt = 0; nt < 4; ++nt)
#pragma unroll
    for (int r = 0; r < 8; ++r) O[nt][r] = 0.f;
  float mloc[8], lloc[8];
#pragma unroll
  for (int r = 0; r < 8; ++r) { mloc[r] = -3.0e38f; lloc[r] = 0.f; }

  for (int j0 = 0; j0 < 1024; j0 += 64) {
    __syncthreads();  // previous tile fully consumed
    {  // K tile: 4x async b128 DMA per thread (ASYNCcnt path)
      int n = tid >> 1, d0 = (tid & 1) * 32;
      const __bf16* kp = Khead + (size_t)(j0 + n) * 64 + d0;
      __bf16* kd = Ks + n * 72 + d0;
#pragma unroll
      for (int c = 0; c < 4; ++c) async_b128(kd + c * 8, kp + c * 8);
    }
    {  // V tile transposed: pack key-pair -> one b32 store
      int np = tid & 31, dh = tid >> 5, d0 = dh * 16;
      const unsigned short* v0 =
          (const unsigned short*)(Vhead + (size_t)(j0 + 2 * np) * 64 + d0);
      const unsigned short* v1 = v0 + 64;
#pragma unroll
      for (int t = 0; t < 16; ++t)
        Vs32[(d0 + t) * 36 + np] = (unsigned)v0[t] | ((unsigned)v1[t] << 16);
    }
    asm volatile("s_wait_asynccnt 0" ::: "memory");
    __syncthreads();

    // logits: 16x64 per wave = 4 n-tiles x 2 k-steps of WMMA
    v8f Sfr[4];
#pragma unroll
    for (int nt = 0; nt < 4; ++nt) {
#pragma unroll
      for (int r = 0; r < 8; ++r) Sfr[nt][r] = 0.f;
      v16bf b0 = load_frag_b(Ks + nt * 16 * 72, 72, 0, lane);
      Sfr[nt] = wmma_bf16(qa0, b0, Sfr[nt]);
      v16bf b1 = load_frag_b(Ks + nt * 16 * 72, 72, 32, lane);
      Sfr[nt] = wmma_bf16(qa1, b1, Sfr[nt]);
    }

    // mask + register-resident online softmax
    float mx[8];
#pragma unroll
    for (int r = 0; r < 8; ++r) mx[r] = -3.0e38f;
#pragma unroll
    for (int nt = 0; nt < 4; ++nt)
#pragma unroll
      for (int r = 0; r < 8; ++r) {
        unsigned char mb =
            mask[((size_t)b * 1024 + qr0 + r + hi * 8) * 1024 + j0 + nt * 16 + nc];
        float sv = mb ? Sfr[nt][r] : -3.0e38f;
        Sfr[nt][r] = sv;
        mx[r] = fmaxf(mx[r], sv);
      }
    float alpha[8];
#pragma unroll
    for (int r = 0; r < 8; ++r) {
#pragma unroll
      for (int m = 8; m > 0; m >>= 1) mx[r] = fmaxf(mx[r], __shfl_xor(mx[r], m, 32));
      float mnew = fmaxf(mloc[r], mx[r]);
      alpha[r] = __expf(mloc[r] - mnew);
      mloc[r] = mnew;
    }
    float ps[8];
#pragma unroll
    for (int r = 0; r < 8; ++r) ps[r] = 0.f;
#pragma unroll
    for (int nt = 0; nt < 4; ++nt)
#pragma unroll
      for (int r = 0; r < 8; ++r) {
        float p = __expf(Sfr[nt][r] - mloc[r]);
        ps[r] += p;
        Pb[w][(r + hi * 8) * 72 + nt * 16 + nc] = f2bf(p);
      }
#pragma unroll
    for (int r = 0; r < 8; ++r) {
#pragma unroll
      for (int m = 8; m > 0; m >>= 1) ps[r] += __shfl_xor(ps[r], m, 32);
      lloc[r] = lloc[r] * alpha[r] + ps[r];
    }
    asm volatile("s_wait_dscnt 0" ::: "memory");  // wave-local Pb RAW

    // rescale accumulators, then O += P @ V
#pragma unroll
    for (int nt = 0; nt < 4; ++nt)
#pragma unroll
      for (int r = 0; r < 8; ++r) O[nt][r] *= alpha[r];

    v16bf pa0 = load_frag_a(Pb[w], 72, 0, lane);
    v16bf pa1 = load_frag_a(Pb[w], 72, 32, lane);
    const __bf16* Vsb = (const __bf16*)Vs32;
#pragma unroll
    for (int nt = 0; nt < 4; ++nt) {
      v16bf vb0 = load_frag_b(Vsb + nt * 16 * 72, 72, 0, lane);
      O[nt] = wmma_bf16(pa0, vb0, O[nt]);
      v16bf vb1 = load_frag_b(Vsb + nt * 16 * 72, 72, 32, lane);
      O[nt] = wmma_bf16(pa1, vb1, O[nt]);
    }
  }

  // final 1/l scaling; write bf16 attn_out (B,S,H*D)
#pragma unroll
  for (int nt = 0; nt < 4; ++nt)
#pragma unroll
    for (int r = 0; r < 8; ++r) {
      int row = qr0 + r + hi * 8;
      float val = O[nt][r] / lloc[r];
      attnOut[((size_t)b * 1024 + row) * 1024 + h * 64 + nt * 16 + nc] =
          f2bf(val);
    }
}

// ---------------------------------------------------------------------------
extern "C" void kernel_launch(void* const* d_in, const int* in_sizes, int n_in,
                              void* d_out, int out_size, void* d_ws,
                              size_t ws_size, hipStream_t stream) {
  (void)in_sizes; (void)n_in; (void)out_size; (void)ws_size;
  const float* hidden = (const float*)d_in[0];
  const float* Wq = (const float*)d_in[1];
  const float* Wk = (const float*)d_in[2];
  const float* Wv = (const float*)d_in[3];
  const float* Wo = (const float*)d_in[4];
  const float* qg = (const float*)d_in[5];
  const float* kg = (const float*)d_in[6];
  const int* pos = (const int*)d_in[7];
  const unsigned char* mask = (const unsigned char*)d_in[8];

  char* ws = (char*)d_ws;
  float*  qkv  = (float*)ws;                                   // 32 MB
  __bf16* Qb   = (__bf16*)(ws + (size_t)32 * 1024 * 1024);     //  8 MB
  __bf16* Kb   = (__bf16*)(ws + (size_t)40 * 1024 * 1024);     //  4 MB
  __bf16* Vb   = (__bf16*)(ws + (size_t)44 * 1024 * 1024);     //  4 MB
  __bf16* aout = (__bf16*)(ws + (size_t)48 * 1024 * 1024);     //  8 MB

  gemm_wmma<float><<<dim3(16, 64), 256, 0, stream>>>(
      hidden, Wq, qkv, 2048, 0, 4096, 1024, 1024);
  gemm_wmma<float><<<dim3(8, 64), 256, 0, stream>>>(
      hidden, Wk, qkv, 2048, 1024, 4096, 512, 1024);
  gemm_wmma<float><<<dim3(8, 64), 256, 0, stream>>>(
      hidden, Wv, qkv, 2048, 1536, 4096, 512, 1024);

  norm_rope<<<16384, 256, 0, stream>>>(qkv, qg, kg, pos, Qb, Kb, Vb);

  attn_kernel<<<dim3(16, 64), 128, 0, stream>>>(Qb, Kb, Vb, mask, aout);

  gemm_wmma<__bf16><<<dim3(16, 64), 256, 0, stream>>>(
      aout, Wo, (float*)d_out, 1024, 0, 4096, 1024, 1024);
}
